// MultiHeadAttention_5729486373651
// MI455X (gfx1250) — matrix operand order repaired and checked by reading, not compiled, not run on hardware
//
#include <hip/hip_runtime.h>
#include <hip/hip_bf16.h>

// ---- problem constants (B,S,D,H from reference) ----
#define B_  2
#define S_  2048
#define H_  16
#define HD_ 64
#define D_  1024
#define M_  (B_ * S_)   // 4096 rows of X

// ---- WMMA fragment types (CDNA5 wave32) ----
typedef __attribute__((ext_vector_type(16))) __bf16 bf16x16;
typedef __attribute__((ext_vector_type(8)))  float  f32x8;

union Frag { bf16x16 v; uint4 q[2]; };

__device__ __forceinline__ __bf16 f32_to_bf16(float f) {
  union { float f; unsigned u; } in; in.f = f;
  unsigned r = in.u + 0x7FFFu + ((in.u >> 16) & 1u);   // RNE
  union { unsigned short s; __bf16 b; } out;
  out.s = (unsigned short)(r >> 16);
  return out.b;
}

__device__ __forceinline__ f32x8 wmma_bf16(const Frag& a, const Frag& b, f32x8 c) {
  // D(16x16 f32) = A(16x32 bf16) x B(32x16 bf16) + C
  return __builtin_amdgcn_wmma_f32_16x16x32_bf16(
      /*neg_a=*/false, a.v, /*neg_b=*/false, b.v,
      /*c_mod=*/(short)0, c, /*reuse_a=*/false, /*reuse_b=*/false);
}

// ---------------------------------------------------------------------------
// fp32 -> bf16 elementwise
__global__ void cvt_f32_to_bf16(const float* __restrict__ in, __bf16* __restrict__ out, int n) {
  int i = blockIdx.x * blockDim.x + threadIdx.x;
  if (i < n) out[i] = f32_to_bf16(in[i]);
}

// W[K,N] fp32 -> Wt[N,K] bf16 (transpose so GEMM B-fragments are contiguous)
__global__ void transpose_to_bf16(const float* __restrict__ w, __bf16* __restrict__ wt) {
  int gid = blockIdx.x * blockDim.x + threadIdx.x;   // D_*D_ threads
  int k = gid >> 10;          // D_ == 1024
  int n = gid & (D_ - 1);
  wt[(size_t)n * D_ + k] = f32_to_bf16(w[gid]);
}

// ---------------------------------------------------------------------------
// C[M,N] = A[M,K](bf16) * Bt[N,K]^T(bf16) + bias, f32 accumulate.
// Each wave computes a 64x32 tile (4x2 WMMA fragments -> 8 WMMAs per 12 b128
// loads); 4 waves arranged 2x2 -> 128x64 block tile.
// mode 0: out bf16 as [B,H,S,HD]   (Q, K)
// mode 1: out bf16 as [B,H,HD,S]   (V transposed for the PV matmul)
// mode 2: out f32  as [M,N]        (final projection -> d_out)
__global__ void gemm_bf16_wmma(const __bf16* __restrict__ A, const __bf16* __restrict__ Bt,
                               const float* __restrict__ bias, void* __restrict__ out,
                               int Kdim, int Ndim, int mode) {
  const int lane = threadIdx.x & 31;
  const int wave = threadIdx.x >> 5;
  const int h    = lane >> 4;     // half-wave select (ISA fragment layouts)
  const int ln   = lane & 15;
  const int m0 = blockIdx.x * 128 + (wave >> 1) * 64;   // 4 M-fragments
  const int n0 = blockIdx.y * 64  + (wave & 1) * 32;    // 2 N-fragments

  f32x8 acc[4][2] = {};

  auto kstep = [&](int k0, bool pf) {
    Frag af[4], bfr[2];
#pragma unroll
    for (int mi = 0; mi < 4; ++mi) {
      const __bf16* p = A + (size_t)(m0 + mi * 16 + ln) * Kdim + k0;
      // A 16x32 bf16 layout: elements -> K = 16h + e (contiguous)
      af[mi].q[0] = *(const uint4*)(p + 16 * h);
      af[mi].q[1] = *(const uint4*)(p + 16 * h + 8);
      if (pf) __builtin_prefetch(p + 128, 0, 1);   // 2 iters ahead; speculative
    }
#pragma unroll
    for (int ni = 0; ni < 2; ++ni) {
      // B 32x16 bf16 layout: column = ln, elements -> K = 16h + e (contiguous)
      const __bf16* p = Bt + (size_t)(n0 + ni * 16 + ln) * Kdim + k0 + 16 * h;
      bfr[ni].q[0] = *(const uint4*)(p);
      bfr[ni].q[1] = *(const uint4*)(p + 8);
      if (pf) __builtin_prefetch(p + 128, 0, 1);
    }
#pragma unroll
    for (int mi = 0; mi < 4; ++mi)
#pragma unroll
      for (int ni = 0; ni < 2; ++ni)
        acc[mi][ni] = wmma_bf16(af[mi], bfr[ni], acc[mi][ni]);
  };

  // manual unroll by 2: prefetch only in the even step, no guard branches
  for (int k0 = 0; k0 < Kdim; k0 += 64) {
    kstep(k0, true);
    kstep(k0 + 32, false);
  }

  // C/D layout: VGPR i holds (M = i + 8h, N = ln)
#pragma unroll
  for (int mi = 0; mi < 4; ++mi)
#pragma unroll
    for (int ni = 0; ni < 2; ++ni)
#pragma unroll
      for (int i = 0; i < 8; ++i) {
        const int m = m0 + mi * 16 + i + 8 * h;
        const int n = n0 + ni * 16 + ln;
        const float v = acc[mi][ni][i] + bias[n];
        if (mode == 2) {
          ((float*)out)[(size_t)m * Ndim + n] = v;
        } else {
          const int bb = m / S_, s = m % S_;
          const int head = n / HD_, hd = n % HD_;
          const size_t idx = (mode == 0)
              ? ((size_t)((bb * H_ + head) * S_ + s) * HD_ + hd)
              : ((size_t)((bb * H_ + head) * HD_ + hd) * S_ + s);
          ((__bf16*)out)[idx] = f32_to_bf16(v);
        }
      }
}

// ---------------------------------------------------------------------------
// Causal flash attention, one wave per 16 query rows.
// Qh/Kh: [B,H,S,HD] bf16 ; VhT: [B,H,HD,S] bf16 ; O: [B,S,D] bf16
__global__ void flash_attn_wmma(const __bf16* __restrict__ Qh, const __bf16* __restrict__ Kh,
                                const __bf16* __restrict__ VhT, __bf16* __restrict__ O) {
  const int lane = threadIdx.x & 31;
  const int wave = threadIdx.x >> 5;
  const int h    = lane >> 4;
  const int ln   = lane & 15;
  const int q0   = blockIdx.x * 64 + wave * 16;
  const int bh   = blockIdx.y;
  const int bb   = bh / H_, head = bh % H_;

  const __bf16* Qb = Qh  + (size_t)bh * S_ * HD_;
  const __bf16* Kb = Kh  + (size_t)bh * S_ * HD_;
  const __bf16* Vb = VhT + (size_t)bh * HD_ * S_;

  // per-wave P tile (16 q x 32 k), row stride 40 bf16 = 80B (16B aligned rows)
  __shared__ __align__(16) __bf16 plds[4][16][40];

  // hoist Q fragments: two K-splits over head_dim 64
  Frag aq[2];
  {
    const __bf16* qp = Qb + (size_t)(q0 + ln) * HD_;
#pragma unroll
    for (int s = 0; s < 2; ++s) {
      aq[s].q[0] = *(const uint4*)(qp + 32 * s + 16 * h);
      aq[s].q[1] = *(const uint4*)(qp + 32 * s + 16 * h + 8);
    }
  }

  float m_i[8], l_i[8];
  f32x8 o[4] = {};
#pragma unroll
  for (int i = 0; i < 8; ++i) { m_i[i] = -1e30f; l_i[i] = 0.f; }
  const float scale = 0.125f;   // 1/sqrt(HD)

  for (int k0 = 0; k0 <= q0 + 15; k0 += 32) {    // causal: only chunks with keys <= last q row
    // ---- S = Q K^T : two 16-key tiles, HD split into 2 wmma K-steps ----
    f32x8 st[2];
#pragma unroll
    for (int t = 0; t < 2; ++t) {
      Frag bk[2];
      const __bf16* kp = Kb + (size_t)(k0 + 16 * t + ln) * HD_;
#pragma unroll
      for (int s = 0; s < 2; ++s) {
        bk[s].q[0] = *(const uint4*)(kp + 32 * s + 16 * h);
        bk[s].q[1] = *(const uint4*)(kp + 32 * s + 16 * h + 8);
      }
      f32x8 sacc = {};
      sacc = wmma_bf16(aq[0], bk[0], sacc);
      sacc = wmma_bf16(aq[1], bk[1], sacc);
      st[t] = sacc;
    }

    // ---- mask + online softmax (rows i+8h live across the 16 lanes of a half) ----
#pragma unroll
    for (int i = 0; i < 8; ++i) {
      const int qg  = q0 + i + 8 * h;
      const int kg0 = k0 + ln;
      const int kg1 = k0 + 16 + ln;
      float s0 = (kg0 <= qg) ? st[0][i] * scale : -1e30f;
      float s1 = (kg1 <= qg) ? st[1][i] * scale : -1e30f;
      float rowm = fmaxf(s0, s1);
#pragma unroll
      for (int off = 1; off < 16; off <<= 1)
        rowm = fmaxf(rowm, __shfl_xor(rowm, off, 32));
      const float m_new = fmaxf(m_i[i], rowm);
      const float alpha = __expf(m_i[i] - m_new);
      const float p0 = (kg0 <= qg) ? __expf(s0 - m_new) : 0.f;
      const float p1 = (kg1 <= qg) ? __expf(s1 - m_new) : 0.f;
      float rsum = p0 + p1;
#pragma unroll
      for (int off = 1; off < 16; off <<= 1)
        rsum += __shfl_xor(rsum, off, 32);
      l_i[i] = l_i[i] * alpha + rsum;
      m_i[i] = m_new;
#pragma unroll
      for (int g = 0; g < 4; ++g) o[g][i] *= alpha;
      // C-fragment -> LDS (row-major P tile) for A-fragment re-layout
      plds[wave][i + 8 * h][ln]      = f32_to_bf16(p0);
      plds[wave][i + 8 * h][ln + 16] = f32_to_bf16(p1);
    }
    asm volatile("s_wait_dscnt 0x0" ::: "memory");   // wave-local LDS RAW fence

    // ---- O += P * V : A-fragment from LDS, B-fragments from V^T (contiguous) ----
    Frag ap;
    {
      const __bf16* pp = &plds[wave][ln][0];
      ap.q[0] = *(const uint4*)(pp + 16 * h);
      ap.q[1] = *(const uint4*)(pp + 16 * h + 8);
    }
#pragma unroll
    for (int g = 0; g < 4; ++g) {
      Frag bv;
      const __bf16* vp = Vb + (size_t)(g * 16 + ln) * S_ + k0 + 16 * h;
      bv.q[0] = *(const uint4*)(vp);
      bv.q[1] = *(const uint4*)(vp + 8);
      o[g] = wmma_bf16(ap, bv, o[g]);
    }
  }

  // ---- normalize and write attn output as [B,S,D] bf16 ----
#pragma unroll
  for (int i = 0; i < 8; ++i) {
    const float inv = 1.0f / l_i[i];
    const int qg = q0 + i + 8 * h;
    const size_t base = ((size_t)bb * S_ + qg) * D_ + (size_t)head * HD_;
#pragma unroll
    for (int g = 0; g < 4; ++g)
      O[base + g * 16 + ln] = f32_to_bf16(o[g][i] * inv);
  }
}

// ---------------------------------------------------------------------------
extern "C" void kernel_launch(void* const* d_in, const int* in_sizes, int n_in,
                              void* d_out, int out_size, void* d_ws, size_t ws_size,
                              hipStream_t stream) {
  const float* X  = (const float*)d_in[0];
  const float* Wq = (const float*)d_in[1];
  const float* bq = (const float*)d_in[2];
  const float* Wk = (const float*)d_in[3];
  const float* bk = (const float*)d_in[4];
  const float* Wv = (const float*)d_in[5];
  const float* bv = (const float*)d_in[6];
  const float* Wo = (const float*)d_in[7];
  const float* bo = (const float*)d_in[8];
  float* out = (float*)d_out;

  char* ws = (char*)d_ws;
  const size_t MB = 1u << 20;
  __bf16* Xbf  = (__bf16*)(ws + 0);        //  8 MiB  X as bf16 [4096,1024]
  __bf16* WqT  = (__bf16*)(ws + 8  * MB);  //  2 MiB  each, transposed bf16
  __bf16* WkT  = (__bf16*)(ws + 10 * MB);
  __bf16* WvT  = (__bf16*)(ws + 12 * MB);
  __bf16* WoT  = (__bf16*)(ws + 14 * MB);
  __bf16* Qh   = (__bf16*)(ws + 16 * MB);  //  8 MiB  [B,H,S,HD]
  __bf16* Kh   = (__bf16*)(ws + 24 * MB);  //  8 MiB  [B,H,S,HD]
  __bf16* VhT  = (__bf16*)(ws + 32 * MB);  //  8 MiB  [B,H,HD,S]
  __bf16* AOut = (__bf16*)(ws + 40 * MB);  //  8 MiB  [B,S,D]
  (void)in_sizes; (void)n_in; (void)out_size; (void)ws_size;

  const int nX = M_ * D_;
  cvt_f32_to_bf16<<<(nX + 255) / 256, 256, 0, stream>>>(X, Xbf, nX);
  transpose_to_bf16<<<(D_ * D_) / 256, 256, 0, stream>>>(Wq, WqT);
  transpose_to_bf16<<<(D_ * D_) / 256, 256, 0, stream>>>(Wk, WkT);
  transpose_to_bf16<<<(D_ * D_) / 256, 256, 0, stream>>>(Wv, WvT);
  transpose_to_bf16<<<(D_ * D_) / 256, 256, 0, stream>>>(Wo, WoT);

  dim3 ggrid(M_ / 128, D_ / 64);         // (32, 16), 128 threads = 4 waves
  gemm_bf16_wmma<<<ggrid, 128, 0, stream>>>(Xbf, WqT, bq, Qh,  D_, D_, 0);
  gemm_bf16_wmma<<<ggrid, 128, 0, stream>>>(Xbf, WkT, bk, Kh,  D_, D_, 0);
  gemm_bf16_wmma<<<ggrid, 128, 0, stream>>>(Xbf, WvT, bv, VhT, D_, D_, 1);

  dim3 fgrid(S_ / 64, B_ * H_);          // (32, 32), 128 threads = 4 waves
  flash_attn_wmma<<<fgrid, 128, 0, stream>>>(Qh, Kh, VhT, AOut);

  gemm_bf16_wmma<<<ggrid, 128, 0, stream>>>(AOut, WoT, bo, out, D_, D_, 2);
}
